// GRUTridentDecoderAttn_2362232013339
// MI455X (gfx1250) — compile-verified, matrix operand order checked by reading, not run on hardware
//
#include <hip/hip_runtime.h>
#include <hip/hip_bf16.h>
#include <stdint.h>

#define TDEPTH 9
#define HDIM 512
#define VDIM 512
#define BDIM 64
#define SDIM 64

typedef __bf16 bf16;
typedef __attribute__((ext_vector_type(16))) __bf16 v16bf;
typedef __attribute__((ext_vector_type(8))) float f32x8;

union FragU {
  v16bf v;
  uint4 q[2];
};

// Load a 16(row) x 32(k) bf16 fragment from a row-major matrix.
// Lane pattern per CDNA5 ISA 7.12.2: lane&15 = row, (lane>>4)*8 = k base,
// VGPRs 0..3 hold k..k+7, VGPRs 4..7 hold k+16..k+23.
__device__ __forceinline__ v16bf load_frag_bf16(const bf16* base, int ld, int lane) {
  const int r = lane & 15;
  const int kb = (lane >> 4) * 8;
  const bf16* p = base + (size_t)r * ld + kb;
  FragU f;
  f.q[0] = *reinterpret_cast<const uint4*>(p);
  f.q[1] = *reinterpret_cast<const uint4*>(p + 16);
  return f.v;
}

__device__ __forceinline__ f32x8 wmma_bf16(v16bf a, v16bf b, f32x8 c) {
  return __builtin_amdgcn_wmma_f32_16x16x32_bf16(false, a, false, b, (short)0, c,
                                                 false, false);
}

__device__ __forceinline__ float sigmoidf_(float x) {
  return 1.0f / (1.0f + __expf(-x));
}

// Preorder position of node (depth d, in-level pos) in the full binary tree.
__device__ __forceinline__ int preorder_index(int d, int pos) {
  int P = 0;
  for (int k = 1; k <= TDEPTH; ++k) {
    if (k > d) break;
    int bit = (pos >> (d - k)) & 1;
    P += 1 + bit * ((1 << (TDEPTH - k + 1)) - 1);
  }
  return P;
}

__global__ __launch_bounds__(256) void f32_to_bf16_kernel(
    const float* __restrict__ s, bf16* __restrict__ d, int nelem) {
  int i = blockIdx.x * 256 + threadIdx.x;
  int stride = gridDim.x * 256;
  for (; i < nelem; i += stride) d[i] = (bf16)s[i];
}

// giA[c][b][g] = b_ih[c][g] + ann_emb[b] . w_ih[c][g][0:512]
__global__ __launch_bounds__(256) void gia_kernel(
    const float* __restrict__ ann, const float* __restrict__ w_ih,
    const float* __restrict__ b_ih, float* __restrict__ giA) {
  int idx = blockIdx.x * 256 + threadIdx.x;
  if (idx >= 2 * BDIM * 1536) return;
  int g = idx % 1536;
  int b = (idx / 1536) % BDIM;
  int c = idx / (1536 * BDIM);
  const float* wr = w_ih + ((size_t)c * 1536 + g) * 1024;
  const float* ae = ann + (size_t)b * HDIM;
  float sum = b_ih[c * 1536 + g];
  for (int i = 0; i < HDIM; ++i) sum += ae[i] * wr[i];
  giA[idx] = sum;
}

// ---------------- head part 1: hid = sigmoid(h @ W1^T + b1) ---------------
#define HS_LD 520   // bf16 elems; 1040B rows: 16B aligned, bank offset 4/row

__global__ __launch_bounds__(256) void hid_kernel(
    const float* __restrict__ h, const bf16* __restrict__ W1b,
    const float* __restrict__ b1, bf16* __restrict__ hid) {
  __shared__ __align__(16) bf16 hS[64 * HS_LD];
  const int tid = threadIdx.x;
  const int lane = tid & 31;
  const int w = tid >> 5;
  const int rowBase = blockIdx.x * 64;

  // stage 64 x 512 h tile (fp32 global -> bf16 LDS)
  for (int i = tid; i < 64 * 512; i += 256) {
    int r = i >> 9, c = i & 511;
    hS[r * HS_LD + c] = (bf16)h[(size_t)(rowBase + r) * HDIM + c];
  }
  __syncthreads();

  // 64 col tiles / 8 waves; 4 row subtiles each
#pragma unroll 1
  for (int cg = 0; cg < 8; ++cg) {
    const int ctA = w * 8 + cg;
    f32x8 acc[4];
#pragma unroll
    for (int s = 0; s < 4; ++s) acc[s] = {};
#pragma unroll 1
    for (int kk = 0; kk < 16; ++kk) {
      v16bf bfr = load_frag_bf16(W1b + (size_t)(ctA * 16) * HDIM + kk * 32, HDIM, lane);
#pragma unroll
      for (int s = 0; s < 4; ++s) {
        v16bf a = load_frag_bf16(hS + s * 16 * HS_LD + kk * 32, HS_LD, lane);
        acc[s] = wmma_bf16(a, bfr, acc[s]);
      }
    }
    const int col = ctA * 16 + (lane & 15);
    const float bb = b1[col];
#pragma unroll
    for (int s = 0; s < 4; ++s) {
#pragma unroll
      for (int j = 0; j < 8; ++j) {
        int m = j + ((lane >> 4) << 3);
        hid[(size_t)(rowBase + s * 16 + m) * 1024 + col] =
            (bf16)sigmoidf_(acc[s][j] + bb);
      }
    }
  }
}

// ---------------- head part 2: prods = hid @ W2^T + b2 -> permuted out ----
#define HIDS_LD 1032 // bf16 elems; 2064B rows: 16B aligned, bank offset 4/row

__global__ __launch_bounds__(256) void prods_kernel(
    const bf16* __restrict__ hid, const bf16* __restrict__ W2b,
    const float* __restrict__ b2, float* __restrict__ out, int d) {
  __shared__ __align__(16) bf16 hidS[64 * HIDS_LD];
  const int tid = threadIdx.x;
  const int lane = tid & 31;
  const int w = tid >> 5;
  const int rowBase = blockIdx.x * 64;
  const int n = 1 << d;

  for (int i = tid; i < 64 * 1024; i += 256) {
    int r = i >> 10, c = i & 1023;
    hidS[r * HIDS_LD + c] = hid[(size_t)(rowBase + r) * 1024 + c];
  }
  __syncthreads();

  // 32 col tiles / 8 waves; 4 row subtiles each
#pragma unroll 1
  for (int cg = 0; cg < 4; ++cg) {
    const int ctB = w * 4 + cg;
    f32x8 acc[4];
#pragma unroll
    for (int s = 0; s < 4; ++s) acc[s] = {};
#pragma unroll 1
    for (int kk = 0; kk < 32; ++kk) {
      v16bf bfr = load_frag_bf16(W2b + (size_t)(ctB * 16) * 1024 + kk * 32, 1024, lane);
#pragma unroll
      for (int s = 0; s < 4; ++s) {
        v16bf a = load_frag_bf16(hidS + s * 16 * HIDS_LD + kk * 32, HIDS_LD, lane);
        acc[s] = wmma_bf16(a, bfr, acc[s]);
      }
    }
    const int col = ctB * 16 + (lane & 15);
    const float bb = b2[col];
#pragma unroll
    for (int s = 0; s < 4; ++s) {
#pragma unroll
      for (int j = 0; j < 8; ++j) {
        int m = j + ((lane >> 4) << 3);
        int rg = rowBase + s * 16 + m;
        int node = rg & (n - 1);
        int b = rg >> d;
        int P = preorder_index(d, node);
        out[((size_t)P * BDIM + b) * VDIM + col] = acc[s][j] + bb;
      }
    }
  }
}

// ---------------- GRU: fused gi/gh GEMM + gate nonlinearity ----------------
#define WS_LD 520

__global__ __launch_bounds__(256) void gru_kernel(
    const float* __restrict__ h, const float* __restrict__ weighted,
    const bf16* __restrict__ wihb, const bf16* __restrict__ whhb,
    const float* __restrict__ giA, const float* __restrict__ b_hh,
    float* __restrict__ h_next, int d) {
  __shared__ __align__(16) bf16 wS[64 * WS_LD];
  __shared__ __align__(16) bf16 hS[64 * WS_LD];
  const int tid = threadIdx.x;
  const int lane = tid & 31;
  const int w = tid >> 5;
  const int rowBase = blockIdx.x * 64;
  const int c = blockIdx.y; // child
  const int n = 1 << d;

  for (int i = tid; i < 64 * 512; i += 256) {
    int r = i >> 9, cc = i & 511;
    size_t gidx = (size_t)(rowBase + r) * HDIM + cc;
    wS[r * WS_LD + cc] = (bf16)weighted[gidx];
    hS[r * WS_LD + cc] = (bf16)h[gidx];
  }
  __syncthreads();

  const bf16* wih_c = wihb + (size_t)c * 1536 * 1024;
  const bf16* whh_c = whhb + (size_t)c * 1536 * 512;
  const float* giA_c = giA + (size_t)c * BDIM * 1536;
  const float* bhh_c = b_hh + (size_t)c * 1536;

#pragma unroll 1
  for (int g = 0; g < 4; ++g) {
    const int ct = g * 8 + w; // H col tile 0..31
    f32x8 gi[3][4]; // [gate r/z/n][row subtile]
    f32x8 gh[3][4];
#pragma unroll
    for (int q = 0; q < 3; ++q)
#pragma unroll
      for (int s = 0; s < 4; ++s) { gi[q][s] = {}; gh[q][s] = {}; }

#pragma unroll 1
    for (int kk = 0; kk < 16; ++kk) {
      // gi: x = [ann|weighted]; ann part folded into giA -> use cols 512..1023
      v16bf bi[3], bh[3];
#pragma unroll
      for (int q = 0; q < 3; ++q) {
        bi[q] = load_frag_bf16(
            wih_c + (size_t)(q * 512 + ct * 16) * 1024 + 512 + kk * 32, 1024, lane);
        bh[q] = load_frag_bf16(
            whh_c + (size_t)(q * 512 + ct * 16) * 512 + kk * 32, 512, lane);
      }
#pragma unroll
      for (int s = 0; s < 4; ++s) {
        v16bf aw = load_frag_bf16(wS + s * 16 * WS_LD + kk * 32, WS_LD, lane);
        v16bf ah = load_frag_bf16(hS + s * 16 * WS_LD + kk * 32, WS_LD, lane);
#pragma unroll
        for (int q = 0; q < 3; ++q) {
          gi[q][s] = wmma_bf16(aw, bi[q], gi[q][s]);
          gh[q][s] = wmma_bf16(ah, bh[q], gh[q][s]);
        }
      }
    }

    const int col = ct * 16 + (lane & 15);
    const float bh_r = bhh_c[col];
    const float bh_z = bhh_c[512 + col];
    const float bh_n = bhh_c[1024 + col];
#pragma unroll
    for (int s = 0; s < 4; ++s) {
#pragma unroll
      for (int j = 0; j < 8; ++j) {
        int m = j + ((lane >> 4) << 3);
        int rg = rowBase + s * 16 + m;
        int node = rg & (n - 1);
        int b = rg >> d;
        const float* gia = giA_c + (size_t)b * 1536;
        float v_ir = gi[0][s][j] + gia[col];
        float v_iz = gi[1][s][j] + gia[512 + col];
        float v_in = gi[2][s][j] + gia[1024 + col];
        float v_hr = gh[0][s][j] + bh_r;
        float v_hz = gh[1][s][j] + bh_z;
        float v_hn = gh[2][s][j] + bh_n;
        float r = sigmoidf_(v_ir + v_hr);
        float z = sigmoidf_(v_iz + v_hz);
        float ng = tanhf(v_in + r * v_hn);
        float hold = h[(size_t)rg * HDIM + col];
        float hc = (1.0f - z) * ng + z * hold;
        h_next[((size_t)b * (2 * n) + node * 2 + c) * HDIM + col] = hc;
      }
    }
  }
}

// ---------------- attention: weighted = softmax(h.enc^T).enc ---------------
__global__ __launch_bounds__(256) void attn_fast_kernel(
    const float* __restrict__ h, const float* __restrict__ enc,
    const uint8_t* __restrict__ mask, float* __restrict__ weighted, int d) {
  __shared__ __align__(16) float hF[16 * 516];
  __shared__ __align__(16) float encS[8 * 516];
  __shared__ float sc[16 * 64];
  const int tid = threadIdx.x;
  const int lane = tid & 31;
  const int w = tid >> 5;
  const int r0 = blockIdx.x * 16;
  const int bq = r0 >> d; // n >= 16 -> same batch for whole block

  for (int i = tid; i < 16 * 512; i += 256) {
    int r = i >> 9, cc = i & 511;
    hF[r * 516 + cc] = h[(size_t)(r0 + r) * HDIM + cc];
  }
  __syncthreads();

  // scores: lane = (row, half); each wave covers 8 source positions
  const int srow = lane & 15;
  const int half = lane >> 4;
#pragma unroll 1
  for (int si = 0; si < 8; ++si) {
    const int s = w * 8 + si;
    const float* ep = enc + ((size_t)s * BDIM + bq) * HDIM + half * 256;
    const float* hp = hF + srow * 516 + half * 256;
    float p = 0.f;
    for (int j = 0; j < 256; j += 4) {
      float4 a = *reinterpret_cast<const float4*>(hp + j);
      float4 e = *reinterpret_cast<const float4*>(ep + j);
      p += a.x * e.x + a.y * e.y + a.z * e.z + a.w * e.w;
    }
    p += __shfl_xor(p, 16);
    if (!mask[bq * SDIM + s]) p = -1.0e9f;
    if (half == 0) sc[srow * 64 + s] = p;
  }
  __syncthreads();

  // softmax: 16 threads per row, 4 source positions each
  {
    const int row = tid >> 4;
    const int q = tid & 15;
    float v0 = sc[row * 64 + q * 4 + 0];
    float v1 = sc[row * 64 + q * 4 + 1];
    float v2 = sc[row * 64 + q * 4 + 2];
    float v3 = sc[row * 64 + q * 4 + 3];
    float mx = fmaxf(fmaxf(v0, v1), fmaxf(v2, v3));
    for (int k = 1; k < 16; k <<= 1) mx = fmaxf(mx, __shfl_xor(mx, k));
    float e0 = __expf(v0 - mx), e1 = __expf(v1 - mx);
    float e2 = __expf(v2 - mx), e3 = __expf(v3 - mx);
    float ss = e0 + e1 + e2 + e3;
    for (int k = 1; k < 16; k <<= 1) ss += __shfl_xor(ss, k);
    float inv = 1.0f / ss;
    sc[row * 64 + q * 4 + 0] = e0 * inv;
    sc[row * 64 + q * 4 + 1] = e1 * inv;
    sc[row * 64 + q * 4 + 2] = e2 * inv;
    sc[row * 64 + q * 4 + 3] = e3 * inv;
  }
  __syncthreads();

  // weighted sum with enc staged through LDS in 8-row s-tiles
  const int row = 2 * w + (lane >> 4);
  const int cb = (lane & 15) * 32;
  float acc[32];
#pragma unroll
  for (int j = 0; j < 32; ++j) acc[j] = 0.f;
#pragma unroll 1
  for (int st = 0; st < 8; ++st) {
    for (int i = tid; i < 8 * 512; i += 256) {
      int rr = i >> 9, cc = i & 511;
      encS[rr * 516 + cc] = enc[((size_t)(st * 8 + rr) * BDIM + bq) * HDIM + cc];
    }
    __syncthreads();
#pragma unroll 1
    for (int s8 = 0; s8 < 8; ++s8) {
      float av = sc[row * 64 + st * 8 + s8];
      const float* ep = encS + s8 * 516 + cb;
#pragma unroll
      for (int j = 0; j < 32; ++j) acc[j] += av * ep[j];
    }
    __syncthreads();
  }
  float* wp = weighted + (size_t)(r0 + row) * HDIM + cb;
#pragma unroll
  for (int j = 0; j < 32; ++j) wp[j] = acc[j];
}

__global__ __launch_bounds__(256) void attn_slow_kernel(
    const float* __restrict__ h, const float* __restrict__ enc,
    const uint8_t* __restrict__ mask, float* __restrict__ weighted, int d) {
  const int lane = threadIdx.x & 31;
  const int w = threadIdx.x >> 5;
  const int row = blockIdx.x * 8 + w;
  const int b = row >> d;
  float hreg[16];
  const float* hp = h + (size_t)row * HDIM + lane * 16;
#pragma unroll
  for (int j = 0; j < 16; ++j) hreg[j] = hp[j];
  float aLo = 0.f, aHi = 0.f;
#pragma unroll 1
  for (int s = 0; s < 64; ++s) {
    const float* ep = enc + ((size_t)s * BDIM + b) * HDIM + lane * 16;
    float p = 0.f;
#pragma unroll
    for (int j = 0; j < 16; ++j) p += hreg[j] * ep[j];
    for (int k = 1; k < 32; k <<= 1) p += __shfl_xor(p, k);
    if (!mask[b * SDIM + s]) p = -1.0e9f;
    if (s < 32) {
      if (lane == s) aLo = p;
    } else {
      if (lane == s - 32) aHi = p;
    }
  }
  float mx = fmaxf(aLo, aHi);
  for (int k = 1; k < 32; k <<= 1) mx = fmaxf(mx, __shfl_xor(mx, k));
  float eLo = __expf(aLo - mx), eHi = __expf(aHi - mx);
  float ss = eLo + eHi;
  for (int k = 1; k < 32; k <<= 1) ss += __shfl_xor(ss, k);
  float inv = 1.0f / ss;
  float acc[16];
#pragma unroll
  for (int j = 0; j < 16; ++j) acc[j] = 0.f;
#pragma unroll 1
  for (int s = 0; s < 64; ++s) {
    float av = (s < 32 ? __shfl(eLo, s) : __shfl(eHi, s - 32)) * inv;
    const float* ep = enc + ((size_t)s * BDIM + b) * HDIM + lane * 16;
#pragma unroll
    for (int j = 0; j < 16; ++j) acc[j] += av * ep[j];
  }
  float* wp = weighted + (size_t)row * HDIM + lane * 16;
#pragma unroll
  for (int j = 0; j < 16; ++j) wp[j] = acc[j];
}

extern "C" void kernel_launch(void* const* d_in, const int* in_sizes, int n_in,
                              void* d_out, int out_size, void* d_ws,
                              size_t ws_size, hipStream_t stream) {
  (void)in_sizes; (void)n_in; (void)out_size; (void)ws_size;
  const float* root = (const float*)d_in[0];
  const float* ann = (const float*)d_in[1];
  const float* enc = (const float*)d_in[2];
  const uint8_t* mask = (const uint8_t*)d_in[3];
  const float* w_ih = (const float*)d_in[4];
  const float* w_hh = (const float*)d_in[5];
  const float* b_ih = (const float*)d_in[6];
  const float* b_hh = (const float*)d_in[7];
  const float* W1 = (const float*)d_in[8];
  const float* b1 = (const float*)d_in[9];
  const float* W2 = (const float*)d_in[10];
  const float* b2 = (const float*)d_in[11];
  float* out = (float*)d_out;

  char* ws = (char*)d_ws;
  size_t off = 0;
  auto take = [&](size_t bytes) {
    char* p = ws + off;
    off = (off + bytes + 255) & ~(size_t)255;
    return p;
  };
  bf16* w1b = (bf16*)take((size_t)1024 * 512 * 2);
  bf16* w2b = (bf16*)take((size_t)512 * 1024 * 2);
  bf16* wihb = (bf16*)take((size_t)2 * 1536 * 1024 * 2);
  bf16* whhb = (bf16*)take((size_t)2 * 1536 * 512 * 2);
  float* giA = (float*)take((size_t)2 * BDIM * 1536 * 4);
  float* weighted = (float*)take((size_t)16384 * 512 * 4);
  float* hA = (float*)take((size_t)32768 * 512 * 4);
  float* hB = (float*)take((size_t)32768 * 512 * 4);
  bf16* hid = (bf16*)take((size_t)32768 * 1024 * 2);

  f32_to_bf16_kernel<<<512, 256, 0, stream>>>(W1, w1b, 1024 * 512);
  f32_to_bf16_kernel<<<512, 256, 0, stream>>>(W2, w2b, 512 * 1024);
  f32_to_bf16_kernel<<<2048, 256, 0, stream>>>(w_ih, wihb, 2 * 1536 * 1024);
  f32_to_bf16_kernel<<<1024, 256, 0, stream>>>(w_hh, whhb, 2 * 1536 * 512);
  gia_kernel<<<(2 * BDIM * 1536 + 255) / 256, 256, 0, stream>>>(ann, w_ih, b_ih, giA);
  hipMemcpyAsync(hA, root, (size_t)BDIM * HDIM * sizeof(float),
                 hipMemcpyDeviceToDevice, stream);

  float* hc = hA;
  float* hn = hB;
  for (int d = 0; d <= TDEPTH; ++d) {
    int n = 1 << d;
    int M = BDIM * n;
    hid_kernel<<<M / 64, 256, 0, stream>>>(hc, w1b, b1, hid);
    prods_kernel<<<M / 64, 256, 0, stream>>>(hid, w2b, b2, out, d);
    if (d < TDEPTH) {
      if (n >= 16)
        attn_fast_kernel<<<M / 16, 256, 0, stream>>>(hc, enc, mask, weighted, d);
      else
        attn_slow_kernel<<<M / 8, 256, 0, stream>>>(hc, enc, mask, weighted, d);
      gru_kernel<<<dim3(M / 64, 2), 256, 0, stream>>>(hc, weighted, wihb, whhb,
                                                      giA, b_hh, hn, d);
      float* t = hc; hc = hn; hn = t;
    }
  }
}
